// MultiHeadAttention_85263690760566
// MI455X (gfx1250) — matrix operand order repaired and checked by reading, not compiled, not run on hardware
//
#include <hip/hip_runtime.h>

typedef __bf16 bf16;
typedef unsigned int u32;
typedef unsigned short u16;
typedef __attribute__((ext_vector_type(16))) __bf16 v16bf;
typedef __attribute__((ext_vector_type(8)))  float  v8f;

#define B_    4
#define M_    8192
#define N_    512
#define KVD_  768
#define QD_   1024
#define H_    8
#define D_    128
#define HID_  1024
// reference divides logits by head_dim**-0.5, i.e. multiplies by sqrt(128)
#define SCALE_ 11.313708498984761f

static __device__ __forceinline__ u32 pack2_bf16(float lo, float hi) {
  u16 a = __builtin_bit_cast(u16, (bf16)lo);
  u16 b = __builtin_bit_cast(u16, (bf16)hi);
  return (u32)a | ((u32)b << 16);
}

static __device__ __forceinline__ v8f wmma_bf16(v16bf a, v16bf b, v8f c) {
  // D = A(16x32 bf16) * B(32x16 bf16) + C(16x16 f32)
  return __builtin_amdgcn_wmma_f32_16x16x32_bf16(false, a, false, b, (short)0, c,
                                                 false, false);
}

#define BM 128
#define BN 64
#define BK 32
#define AST 40   // A tile LDS stride [m][k], padded, even
#define WTS 40   // W tile LDS stride [n][k] (transposed), padded, even

// ---- fragment load helpers (ISA 7.12.2 layouts) ---------------------------
static __device__ __forceinline__ v16bf load_afrag(const bf16* row, int lh) {
  // A 16x32 bf16: lane half 0 -> K 0..7,16..23 ; half 1 -> K 8..15,24..31
  const u32* arow = (const u32*)row;
  alignas(32) u32 t8[8];
  const int kb = lh * 8;
  #pragma unroll
  for (int t = 0; t < 4; ++t) {
    t8[t]     = arow[(kb >> 1) + t];
    t8[4 + t] = arow[((kb + 16) >> 1) + t];
  }
  return *(const v16bf*)t8;
}
static __device__ __forceinline__ v16bf load_bfrag(const bf16* row, int lh) {
  // B 32x16 bf16 from k-contiguous storage: half 0 -> K 0..7,16..23 ; half 1 -> K 8..15,24..31
  const u32* wrow = (const u32*)row;
  alignas(32) u32 t8[8];
  const int kb2 = lh * 8;
  #pragma unroll
  for (int t = 0; t < 4; ++t) {
    t8[t]     = wrow[(kb2 >> 1) + t];
    t8[4 + t] = wrow[((kb2 + 16) >> 1) + t];
  }
  return *(const v16bf*)t8;
}

// ---------------------------------------------------------------------------
// Tiled GEMM: C[M x N] = A[M x K] * W[K x N] + bias, bf16 WMMA math.
// Block: 256 threads = 8 waves, tile 128x64x32; wave tile 32x32 = 2x2 WMMA.
// W tile staged TRANSPOSED ([n][k]) so B fragments are contiguous ds_load_b128.
// ---------------------------------------------------------------------------
template <int A_BF16, int OUT_BF16>
__global__ __launch_bounds__(256, 2)
void gemm_wmma_kernel(const void* __restrict__ Ain,
                      const float* __restrict__ W,
                      const float* __restrict__ bias,
                      void* __restrict__ Cout,
                      int K, int N)
{
  __shared__ bf16 As[BM * AST];   // [m][k]
  __shared__ bf16 Wt[BN * WTS];   // [n][k]

  const int tid  = threadIdx.x;
  const int wave = tid >> 5;
  const int lane = tid & 31;
  const int wm = wave >> 1, wn = wave & 1;
  const int lm = lane & 15, lh = lane >> 4;
  const int bm = blockIdx.x * BM;
  const int bn = blockIdx.y * BN;

  v8f acc[2][2] = {};

  for (int k0 = 0; k0 < K; k0 += BK) {
    const int more = (k0 + BK) < K;
    __syncthreads();
    if constexpr (A_BF16) {
      const u32* Ag = (const u32*)((const bf16*)Ain + (size_t)bm * K + k0);
      const int kw = K >> 1;
      for (int t = tid; t < BM * (BK / 2); t += 256) {
        int r = t >> 4, c = t & 15;
        ((u32*)(As + r * AST))[c] = Ag[(size_t)r * kw + c];
      }
    } else {
      const float* Ag = (const float*)Ain + (size_t)bm * K + k0;
      for (int t = tid; t < BM * (BK / 2); t += 256) {
        int r = t >> 4, c = t & 15;
        const float* p = Ag + (size_t)r * K + 2 * c;
        ((u32*)(As + r * AST))[c] = pack2_bf16(p[0], p[1]);
        if (more) __builtin_prefetch(p + BK, 0, 3);
      }
    }
    {
      const float* Wg = W + (size_t)k0 * N + bn;
      for (int t = tid; t < (BK / 2) * BN; t += 256) {
        int r = t >> 6, c = t & 63;            // k pair / n
        const float* p = Wg + (size_t)(2 * r) * N + c;
        ((u32*)(Wt + c * WTS))[r] = pack2_bf16(p[0], p[(size_t)N]);
        if (more) __builtin_prefetch(p + (size_t)BK * N, 0, 3);
      }
    }
    __syncthreads();

    v16bf afr[2], bfr[2];
    #pragma unroll
    for (int mi = 0; mi < 2; ++mi)
      afr[mi] = load_afrag(As + (wm * 32 + mi * 16 + lm) * AST, lh);
    #pragma unroll
    for (int ni = 0; ni < 2; ++ni)
      bfr[ni] = load_bfrag(Wt + (wn * 32 + ni * 16 + lm) * WTS, lh);
    #pragma unroll
    for (int mi = 0; mi < 2; ++mi)
      #pragma unroll
      for (int ni = 0; ni < 2; ++ni)
        acc[mi][ni] = wmma_bf16(afr[mi], bfr[ni], acc[mi][ni]);
  }

  #pragma unroll
  for (int ni = 0; ni < 2; ++ni) {
    const int n = bn + wn * 32 + ni * 16 + lm;
    const float bb = bias[n];
    #pragma unroll
    for (int mi = 0; mi < 2; ++mi) {
      const size_t base = (size_t)(bm + wm * 32 + mi * 16 + lh * 8) * N + n;
      if constexpr (OUT_BF16) {
        bf16* cp = (bf16*)Cout + base;
        #pragma unroll
        for (int i = 0; i < 8; ++i) { *cp = (bf16)(acc[mi][ni][i] + bb); cp += N; }
      } else {
        float* cp = (float*)Cout + base;
        #pragma unroll
        for (int i = 0; i < 8; ++i) { *cp = acc[mi][ni][i] + bb; cp += N; }
      }
    }
  }
}

// ---------------------------------------------------------------------------
// Fused K+V projection GEMM: shares the staged A tile (inputs_kv) between the
// two weight matrices -> halves the dominant activation read stream and does
// 8 WMMAs per k-step against 2 A-frag + 4 B-frag LDS loads.
// ---------------------------------------------------------------------------
__global__ __launch_bounds__(256, 2)
void gemm_kv_wmma_kernel(const float* __restrict__ Ain,
                         const float* __restrict__ Wka,
                         const float* __restrict__ bka,
                         const float* __restrict__ Wva,
                         const float* __restrict__ bva,
                         bf16* __restrict__ Ck, bf16* __restrict__ Cv,
                         int K, int N)
{
  __shared__ bf16 As[BM * AST];    // [m][k]
  __shared__ bf16 Wkt[BN * WTS];   // [n][k]
  __shared__ bf16 Wvt[BN * WTS];   // [n][k]

  const int tid  = threadIdx.x;
  const int wave = tid >> 5;
  const int lane = tid & 31;
  const int wm = wave >> 1, wn = wave & 1;
  const int lm = lane & 15, lh = lane >> 4;
  const int bm = blockIdx.x * BM;
  const int bn = blockIdx.y * BN;

  v8f acck[2][2] = {};
  v8f accv[2][2] = {};

  for (int k0 = 0; k0 < K; k0 += BK) {
    const int more = (k0 + BK) < K;
    __syncthreads();
    {
      const float* Ag = Ain + (size_t)bm * K + k0;
      for (int t = tid; t < BM * (BK / 2); t += 256) {
        int r = t >> 4, c = t & 15;
        const float* p = Ag + (size_t)r * K + 2 * c;
        ((u32*)(As + r * AST))[c] = pack2_bf16(p[0], p[1]);
        if (more) __builtin_prefetch(p + BK, 0, 3);
      }
      const float* Wkg = Wka + (size_t)k0 * N + bn;
      const float* Wvg = Wva + (size_t)k0 * N + bn;
      for (int t = tid; t < (BK / 2) * BN; t += 256) {
        int r = t >> 6, c = t & 63;
        const float* pk = Wkg + (size_t)(2 * r) * N + c;
        const float* pv = Wvg + (size_t)(2 * r) * N + c;
        ((u32*)(Wkt + c * WTS))[r] = pack2_bf16(pk[0], pk[(size_t)N]);
        ((u32*)(Wvt + c * WTS))[r] = pack2_bf16(pv[0], pv[(size_t)N]);
        if (more) {
          __builtin_prefetch(pk + (size_t)BK * N, 0, 3);
          __builtin_prefetch(pv + (size_t)BK * N, 0, 3);
        }
      }
    }
    __syncthreads();

    v16bf afr[2], bkfr[2], bvfr[2];
    #pragma unroll
    for (int mi = 0; mi < 2; ++mi)
      afr[mi] = load_afrag(As + (wm * 32 + mi * 16 + lm) * AST, lh);
    #pragma unroll
    for (int ni = 0; ni < 2; ++ni) {
      bkfr[ni] = load_bfrag(Wkt + (wn * 32 + ni * 16 + lm) * WTS, lh);
      bvfr[ni] = load_bfrag(Wvt + (wn * 32 + ni * 16 + lm) * WTS, lh);
    }
    #pragma unroll
    for (int mi = 0; mi < 2; ++mi)
      #pragma unroll
      for (int ni = 0; ni < 2; ++ni) {
        acck[mi][ni] = wmma_bf16(afr[mi], bkfr[ni], acck[mi][ni]);
        accv[mi][ni] = wmma_bf16(afr[mi], bvfr[ni], accv[mi][ni]);
      }
  }

  #pragma unroll
  for (int ni = 0; ni < 2; ++ni) {
    const int n = bn + wn * 32 + ni * 16 + lm;
    const float bbk = bka[n];
    const float bbv = bva[n];
    #pragma unroll
    for (int mi = 0; mi < 2; ++mi) {
      const size_t base = (size_t)(bm + wm * 32 + mi * 16 + lh * 8) * N + n;
      bf16* ck = Ck + base;
      bf16* cv = Cv + base;
      #pragma unroll
      for (int i = 0; i < 8; ++i) {
        *ck = (bf16)(acck[mi][ni][i] + bbk);  ck += N;
        *cv = (bf16)(accv[mi][ni][i] + bbv);  cv += N;
      }
    }
  }
}

// ---------------------------------------------------------------------------
// Flash attention: per block one (b, h, 64-row q tile); 4 waves x 16 q rows.
// Streams K/V in 64-kv tiles, online softmax, O accumulated in registers.
// ---------------------------------------------------------------------------
#define FQT 64
#define FMT 64
#define KTS 136   // K tile LDS stride [kv][d], 128+8 pad
#define VTS 72    // V tile (transposed) stride [d][kv], 64+8 pad
#define PST 64    // per-wave P staging stride

__global__ __launch_bounds__(128, 1)
void flash_attn_kernel(const bf16* __restrict__ Qb, const bf16* __restrict__ Kb,
                       const bf16* __restrict__ Vb, const int* __restrict__ mask,
                       bf16* __restrict__ Ob)
{
  __shared__ bf16 Kt[FMT * KTS];      // [kv][d]
  __shared__ bf16 Vt[D_ * VTS];       // [d][kv]  (transposed at load)
  __shared__ bf16 Ps[4 * 16 * PST];   // per-wave P (16 x 64)

  const int tid  = threadIdx.x;
  const int wave = tid >> 5;
  const int lane = tid & 31;
  const int lm = lane & 15;
  const int lh = lane >> 4;

  int bid = blockIdx.x;
  const int qt = bid & 7;  bid >>= 3;   // N_/FQT == 8
  const int h  = bid & 7;  bid >>= 3;   // H_ == 8
  const int b  = bid;

  const int q0 = qt * FQT + wave * 16;

  // Q fragments held in registers for the whole kernel (16 rows x 128 dim)
  v16bf qf[4];
  {
    const bf16* qrow = Qb + ((size_t)(b * N_ + q0 + lm)) * HID_ + h * D_;
    #pragma unroll
    for (int kk = 0; kk < 4; ++kk)
      qf[kk] = load_afrag(qrow + kk * 32, lh);
  }

  v8f o[8] = {};            // 16 q rows x 128 head dim accumulator
  float rm[8], rs[8];       // per-row running max / sum (rows 8*lh+i)
  #pragma unroll
  for (int i = 0; i < 8; ++i) { rm[i] = -3.0e38f; rs[i] = 0.0f; }

  const int* maskbase = mask + ((size_t)(b * N_ + q0 + lh * 8)) * M_ + lm;
  const bf16* Kbb = Kb + ((size_t)b * M_) * HID_ + h * D_;
  const bf16* Vbb = Vb + ((size_t)b * M_) * HID_ + h * D_;

  for (int m0 = 0; m0 < M_; m0 += FMT) {
    __syncthreads();
    // ---- cooperative load of K tile (row major) and V tile (transposed) ----
    {
      const u32* Kg = (const u32*)(Kbb + (size_t)m0 * HID_);
      const u32* Vg = (const u32*)(Vbb + (size_t)m0 * HID_);
      const int hw = HID_ >> 1;
      u16* vt = (u16*)Vt;
      for (int t = tid; t < FMT * (D_ / 2); t += 128) {
        int r = t >> 6, c = t & 63;     // r = kv row, c = head-dim pair
        ((u32*)(Kt + r * KTS))[c] = Kg[(size_t)r * hw + c];
        u32 v = Vg[(size_t)r * hw + c];
        vt[(2 * c)     * VTS + r] = (u16)(v & 0xffffu);
        vt[(2 * c + 1) * VTS + r] = (u16)(v >> 16);
      }
      // prefetch next K/V/mask tiles while this one is consumed
      if (m0 + FMT < M_) {
        const char* Kn = (const char*)(Kbb + (size_t)(m0 + FMT) * HID_);
        const char* Vn = (const char*)(Vbb + (size_t)(m0 + FMT) * HID_);
        const char* Mn = (const char*)(mask +
            ((size_t)(b * N_ + qt * FQT)) * M_ + (m0 + FMT));
        const int r = tid & 63, half = tid >> 6;   // 64 rows x 2 cachelines
        __builtin_prefetch(Kn + (size_t)r * (HID_ * 2) + half * 128, 0, 3);
        __builtin_prefetch(Vn + (size_t)r * (HID_ * 2) + half * 128, 0, 3);
        __builtin_prefetch(Mn + (size_t)r * (M_ * 4) + half * 128, 0, 3);
      }
    }
    __syncthreads();

    // ---- S = Q * K^T  (four 16-wide kv sub-tiles) ----
    v8f s[4];
    #pragma unroll
    for (int j = 0; j < 4; ++j) {
      v8f sj = {};
      const bf16* krow = Kt + (16 * j + lm) * KTS;
      #pragma unroll
      for (int kk = 0; kk < 4; ++kk)
        sj = wmma_bf16(qf[kk], load_bfrag(krow + kk * 32, lh), sj);
      s[j] = sj;
    }

    // ---- mask + online softmax ----
    float p[4][8];
    float tmax[8];
    #pragma unroll
    for (int i = 0; i < 8; ++i) tmax[i] = -3.0e38f;
    #pragma unroll
    for (int j = 0; j < 4; ++j) {
      const int* mp = maskbase + m0 + 16 * j;
      #pragma unroll
      for (int i = 0; i < 8; ++i) {
        const int mv = *mp;  mp += M_;
        const float sv = mv ? (s[j][i] * SCALE_) : -3.0e38f;
        p[j][i] = sv;
        tmax[i] = fmaxf(tmax[i], sv);
      }
    }
    #pragma unroll
    for (int i = 0; i < 8; ++i) {
      float v = tmax[i];
      #pragma unroll
      for (int off = 1; off < 16; off <<= 1)
        v = fmaxf(v, __shfl_xor(v, off, 16));
      const float nm = fmaxf(rm[i], v);
      const float corr = __expf(rm[i] - nm);   // 0 if row was empty, 1 if both empty
      float rsum = 0.0f;
      #pragma unroll
      for (int j = 0; j < 4; ++j) {
        const float pv = (p[j][i] <= -1.0e37f) ? 0.0f : __expf(p[j][i] - nm);
        p[j][i] = pv;
        rsum += pv;
      }
      #pragma unroll
      for (int off = 1; off < 16; off <<= 1)
        rsum += __shfl_xor(rsum, off, 16);
      rs[i] = rs[i] * corr + rsum;
      rm[i] = nm;
      #pragma unroll
      for (int jj = 0; jj < 8; ++jj) o[jj][i] *= corr;
    }

    // ---- stage P as bf16 in per-wave LDS (C layout -> memory) ----
    bf16* pw = Ps + wave * 16 * PST;
    #pragma unroll
    for (int j = 0; j < 4; ++j)
      #pragma unroll
      for (int i = 0; i < 8; ++i)
        pw[(lh * 8 + i) * PST + 16 * j + lm] = (bf16)p[j][i];

    // ---- O += P * V  (A from Ps, B from transposed Vt; same-wave LDS in-order)
    #pragma unroll
    for (int kk2 = 0; kk2 < 2; ++kk2) {
      const v16bf af = load_afrag(pw + lm * PST + kk2 * 32, lh);
      #pragma unroll
      for (int jj = 0; jj < 8; ++jj) {
        const v16bf bfrag =
            load_bfrag(Vt + (16 * jj + lm) * VTS + kk2 * 32, lh);
        o[jj] = wmma_bf16(af, bfrag, o[jj]);
      }
    }
  }

  // ---- normalize and store attention output (B, N, HIDDEN) bf16 ----
  float inv[8];
  #pragma unroll
  for (int i = 0; i < 8; ++i) inv[i] = (rs[i] > 0.0f) ? 1.0f / rs[i] : 0.0f;
  #pragma unroll
  for (int jj = 0; jj < 8; ++jj) {
    bf16* op = Ob + ((size_t)(b * N_ + q0 + lh * 8)) * HID_ + h * D_ + 16 * jj + lm;
    #pragma unroll
    for (int i = 0; i < 8; ++i) {
      *op = (bf16)(o[jj][i] * inv[i]);
      op += HID_;
    }
  }
}

// ---------------------------------------------------------------------------
extern "C" void kernel_launch(void* const* d_in, const int* in_sizes, int n_in,
                              void* d_out, int out_size, void* d_ws, size_t ws_size,
                              hipStream_t stream)
{
  const float* inputs_kv = (const float*)d_in[0];
  const float* inputs_q  = (const float*)d_in[1];
  const int*   amask     = (const int*)  d_in[2];
  const float* Wk = (const float*)d_in[3];
  const float* bk = (const float*)d_in[4];
  const float* Wq = (const float*)d_in[5];
  const float* bq = (const float*)d_in[6];
  const float* Wv = (const float*)d_in[7];
  const float* bv = (const float*)d_in[8];
  const float* Wp = (const float*)d_in[9];
  const float* bp = (const float*)d_in[10];
  float* out = (float*)d_out;
  (void)in_sizes; (void)n_in; (void)out_size; (void)ws_size;

  char* ws = (char*)d_ws;
  size_t off = 0;
  auto take = [&](size_t bytes) -> void* {
    void* p = ws + off;
    off += (bytes + 255) & ~(size_t)255;
    return p;
  };
  bf16* Kbuf = (bf16*)take((size_t)B_ * M_ * HID_ * sizeof(bf16));  // 64 MB
  bf16* Vbuf = (bf16*)take((size_t)B_ * M_ * HID_ * sizeof(bf16));  // 64 MB
  bf16* Qbuf = (bf16*)take((size_t)B_ * N_ * HID_ * sizeof(bf16));  //  4 MB
  bf16* Abuf = (bf16*)take((size_t)B_ * N_ * HID_ * sizeof(bf16));  //  4 MB

  // fused K+V projections: (B*M x KVD) @ (KVD x HID) + bias -> bf16 (A read once)
  dim3 gKV((B_ * M_) / BM, HID_ / BN);
  gemm_kv_wmma_kernel<<<gKV, 256, 0, stream>>>(inputs_kv, Wk, bk, Wv, bv,
                                               Kbuf, Vbuf, KVD_, HID_);
  // Q projection: (B*N x QD) @ (QD x HID) + bias -> bf16
  dim3 gQ((B_ * N_) / BM, HID_ / BN);
  gemm_wmma_kernel<0, 1><<<gQ, 256, 0, stream>>>(inputs_q, Wq, bq, Qbuf,
                                                 QD_, HID_);
  // fused attention (softmax streaming over M)
  flash_attn_kernel<<<B_ * H_ * (N_ / FQT), 128, 0, stream>>>(Qbuf, Kbuf, Vbuf,
                                                              amask, Abuf);
  // output projection: (B*N x HID) @ (HID x QD) + bias -> fp32 d_out
  gemm_wmma_kernel<1, 0><<<gQ, 256, 0, stream>>>(Abuf, Wp, bp, out,
                                                 HID_, QD_);
}
